// MultiHeadAttention_25082609009200
// MI455X (gfx1250) — matrix operand order repaired and checked by reading, not compiled, not run on hardware
//
#include <hip/hip_runtime.h>
#include <hip/hip_bf16.h>
#include <stdint.h>

#define HD 64
#define NH 16
#define MD 1024
#define SQ 2048
#define BA 2

typedef _Float16 v16h __attribute__((ext_vector_type(16)));
typedef _Float16 v8h  __attribute__((ext_vector_type(8)));
typedef float    v8f  __attribute__((ext_vector_type(8)));
typedef float    v4f  __attribute__((ext_vector_type(4)));
typedef unsigned int v4u __attribute__((ext_vector_type(4)));
typedef int      v8i  __attribute__((ext_vector_type(8)));
typedef int      v4i  __attribute__((ext_vector_type(4)));

#if defined(__has_builtin)
#if __has_builtin(__builtin_amdgcn_tensor_load_to_lds)
#define HAVE_TDM 1
#endif
#endif
#ifndef HAVE_TDM
#define HAVE_TDM 0
#endif

static __device__ __forceinline__ v16h cat16(v8h lo, v8h hi) {
  v16h r;
#pragma unroll
  for (int i = 0; i < 8; ++i) { r[i] = lo[i]; r[i + 8] = hi[i]; }
  return r;
}
static __device__ __forceinline__ v8h ld8(const _Float16* p) {
  return *reinterpret_cast<const v8h*>(p);
}
static __device__ __forceinline__ v8h cvt8(const float* p) {
  v4f a = *reinterpret_cast<const v4f*>(p);
  v4f b = *reinterpret_cast<const v4f*>(p + 4);
  v8h r;
#pragma unroll
  for (int i = 0; i < 4; ++i) { r[i] = (_Float16)a[i]; r[i + 4] = (_Float16)b[i]; }
  return r;
}
static __device__ __forceinline__ v8h cvt8(const _Float16* p) { return ld8(p); }

// ---- 16-lane butterfly reductions via DPP16 (pure VALU, no LDS traffic) ----
template <int CTRL>
static __device__ __forceinline__ float dppf(float x) {
  int r = __builtin_amdgcn_update_dpp(0, __builtin_bit_cast(int, x), CTRL,
                                      0xf, 0xf, true);
  return __builtin_bit_cast(float, r);
}
static __device__ __forceinline__ float row16_max(float v) {
  v = fmaxf(v, dppf<0xB1>(v));   // quad_perm [1,0,3,2]
  v = fmaxf(v, dppf<0x4E>(v));   // quad_perm [2,3,0,1]
  v = fmaxf(v, dppf<0x141>(v));  // row_half_mirror
  v = fmaxf(v, dppf<0x140>(v));  // row_mirror
  return v;
}
static __device__ __forceinline__ float row16_sum(float v) {
  v += dppf<0xB1>(v);
  v += dppf<0x4E>(v);
  v += dppf<0x141>(v);
  v += dppf<0x140>(v);
  return v;
}

#if HAVE_TDM
// ---------------------------------------------------------------------------
// TDM: DMA one 3D tile (X=64 halves contiguous, Y=16 rows @ stride 192,
// Z=2 rows @ stride 3072) into LDS, padded 4 DWORDs after every 32 DWORDs
// so each 64-half row lands at a 72-half stride (matches fragment readers).
// D# packing per CDNA5 ISA §8.3-8.6.
// ---------------------------------------------------------------------------
static __device__ __forceinline__ void tdm_load_tile(uint32_t lds_addr,
                                                     const void* gptr) {
  uint64_t ga = (uint64_t)(uintptr_t)gptr;
  // Group 0: count=1 | lds_addr | global_addr[56:0] | type=2
  v4u g0 = {1u, lds_addr, (uint32_t)ga, (uint32_t)(ga >> 32) | (2u << 30)};
  // Group 1: data_size=2B, pad_enable, pad_interval=4 (32 dw), pad_amount=3 (4 dw)
  uint32_t w0 = (1u << 16) | (1u << 20) | (4u << 22) | (3u << 25);
  const uint32_t dim0 = 3072u;       // tensor_dim0 (>= tile X extent)
  const uint32_t dim1 = (1u << 20);  // tensor_dim1 (large, no OOB)
  uint32_t w1 = (dim0 & 0xFFFFu) << 16;                  // atomic_addr=0 | dim0.lo
  uint32_t w2 = (dim0 >> 16) | ((dim1 & 0xFFFFu) << 16); // dim0.hi | dim1.lo
  uint32_t w3 = (dim1 >> 16) | (64u << 16);              // dim1.hi | tile_dim0=64
  uint32_t w4 = 16u | (2u << 16);                        // tile_dim1=16 | tile_dim2=2
  uint32_t w5 = 192u;                                    // tensor_dim0_stride.lo
  uint32_t w6 = 0u | ((3072u & 0xFFFFu) << 16);          // stride0.hi | stride1.lo
  uint32_t w7 = 3072u >> 16;                             // stride1.hi
  v8i g1 = {(int)w0, (int)w1, (int)w2, (int)w3,
            (int)w4, (int)w5, (int)w6, (int)w7};
  // Group 2: tensor_dim2 large; tile_dim3=0 (unused)
  v4i g2 = {(int)(1u << 20), 0, 0, 0};
  v4i g3 = {0, 0, 0, 0};
#if __clang_major__ >= 23
  v8i z8 = {0, 0, 0, 0, 0, 0, 0, 0};
  __builtin_amdgcn_tensor_load_to_lds(g0, g1, g2, g3, z8, 0);
#else
  __builtin_amdgcn_tensor_load_to_lds(g0, g1, g2, g3, 0);
#endif
}
#endif

// ---------------------------------------------------------------------------
// Tiled GEMM:  C[M,N] = cast(A)[M,K] @ cast(W)[K,N] + bias[N]
// f16 operands, f32 WMMA accumulate. Block: 256 thr = 8 waves, 128x64 C tile.
// ---------------------------------------------------------------------------
template <typename TA, typename TOUT>
__global__ __launch_bounds__(256) void gemm_bias_kernel(
    const TA* __restrict__ A, const float* __restrict__ W,
    const float* __restrict__ bias, TOUT* __restrict__ C,
    int M, int N, int K) {
  const int LDT = 40;
  __shared__ _Float16 lA[128 * 40];  // [m][k]
  __shared__ _Float16 lB[64 * 40];   // transposed: [n][k]
  const int tid = threadIdx.x;
  const int lane = tid & 31, wave = tid >> 5;
  const int half = lane >> 4, l16 = lane & 15;
  const int m0 = blockIdx.y * 128, n0 = blockIdx.x * 64;
  const int waveM = (wave & 3) * 32, waveN = (wave >> 2) * 32;

  const int ar = tid >> 1, akc = (tid & 1) * 16;  // A: 128 rows x 32 k
  const int bkk = tid >> 3, bn = (tid & 7) * 8;   // B: 32 k x 64 n

  v8f c00 = {}, c01 = {}, c10 = {}, c11 = {};

  for (int k0 = 0; k0 < K; k0 += 32) {
    {
      const TA* pA = &A[(size_t)(m0 + ar) * K + k0 + akc];
      *reinterpret_cast<v8h*>(&lA[ar * LDT + akc]) = cvt8(pA);
      *reinterpret_cast<v8h*>(&lA[ar * LDT + akc + 8]) = cvt8(pA + 8);
      if (k0 + 32 < K) __builtin_prefetch(pA + 32, 0, 3);
    }
    {
      const float* pW = &W[(size_t)(k0 + bkk) * N + n0 + bn];
      v8h w8 = cvt8(pW);
#pragma unroll
      for (int j = 0; j < 8; ++j) lB[(bn + j) * LDT + bkk] = w8[j];
      if (k0 + 32 < K) __builtin_prefetch(pW + (size_t)32 * N, 0, 3);
    }
    __syncthreads();

    // All fragments first, then WMMAs back-to-back (single dscnt wait).
    const _Float16* pa0 = &lA[(waveM + l16) * LDT];
    const _Float16* pa1 = &lA[(waveM + 16 + l16) * LDT];
    v16h a0 = cat16(ld8(pa0 + half * 16), ld8(pa0 + half * 16 + 8));
    v16h a1 = cat16(ld8(pa1 + half * 16), ld8(pa1 + half * 16 + 8));
    const _Float16* pb0 = &lB[(waveN + l16) * LDT + half * 16];
    const _Float16* pb1 = &lB[(waveN + 16 + l16) * LDT + half * 16];
    v16h b0 = cat16(ld8(pb0), ld8(pb0 + 8));
    v16h b1 = cat16(ld8(pb1), ld8(pb1 + 8));

    c00 = __builtin_amdgcn_wmma_f32_16x16x32_f16(false, a0, false, b0, (short)0, c00, false, false);
    c01 = __builtin_amdgcn_wmma_f32_16x16x32_f16(false, a0, false, b1, (short)0, c01, false, false);
    c10 = __builtin_amdgcn_wmma_f32_16x16x32_f16(false, a1, false, b0, (short)0, c10, false, false);
    c11 = __builtin_amdgcn_wmma_f32_16x16x32_f16(false, a1, false, b1, (short)0, c11, false, false);
    __syncthreads();
  }

  const int col0 = n0 + waveN + l16;
  const int col1 = col0 + 16;
  const float b0v = bias[col0], b1v = bias[col1];
#pragma unroll
  for (int i = 0; i < 8; ++i) {
    int row0 = m0 + waveM + i + half * 8;
    int row1 = row0 + 16;
    C[(size_t)row0 * N + col0] = (TOUT)(c00[i] + b0v);
    C[(size_t)row0 * N + col1] = (TOUT)(c01[i] + b1v);
    C[(size_t)row1 * N + col0] = (TOUT)(c10[i] + b0v);
    C[(size_t)row1 * N + col1] = (TOUT)(c11[i] + b1v);
  }
}

// ---------------------------------------------------------------------------
// Flash attention; K/V tiles staged to LDS by the Tensor Data Mover.
// qkv buffer laid out (B, S, 3072); the reference's reshape puts the 192-chan
// chunk for (h,s) at row h*128 + s/16, col (s%16)*192 (kv0%16==0 -> col 0).
// ---------------------------------------------------------------------------
__global__ __launch_bounds__(256) void attn_kernel(
    const _Float16* __restrict__ qkv, _Float16* __restrict__ O) {
  __shared__ _Float16 lK[32 * 72];     // [kv_row][d]  stride 72 halves
  __shared__ _Float16 lV[32 * 72];     // [kv_row][d]  stride 72 halves
  __shared__ _Float16 lP[8][16 * 40];  // per-wave P: [m][kv] stride 40

  const int tid = threadIdx.x;
  const int lane = tid & 31, wave = tid >> 5;
  const int half = lane >> 4, l16 = lane & 15;
  const int b = blockIdx.z, h = blockIdx.y;
  const int qbase = blockIdx.x * 128 + wave * 16;
  const size_t batchOff = (size_t)b * SQ * 3072;

  // ---- q fragments for this wave (scaled by 1/sqrt(64) = 1/8) ----
  v16h q0, q1;
  {
    int s = qbase + l16;
    int rowg = h * 128 + (s >> 4);
    int coloff = (s & 15) * 192;
    const _Float16* qp = qkv + batchOff + (size_t)rowg * 3072 + coloff;
    int d0 = half * 16;
    v8h c0 = ld8(qp + d0), c1 = ld8(qp + 8 + d0);
    v8h c2 = ld8(qp + 32 + d0), c3 = ld8(qp + 40 + d0);
    const _Float16 sc = (_Float16)0.125f;
#pragma unroll
    for (int i = 0; i < 8; ++i) { c0[i] *= sc; c1[i] *= sc; c2[i] *= sc; c3[i] *= sc; }
    q0 = cat16(c0, c1);
    q1 = cat16(c2, c3);
  }

  v8f oacc[4] = {};
  float mrun[8], lrun[8];
#pragma unroll
  for (int i = 0; i < 8; ++i) { mrun[i] = -1e30f; lrun[i] = 0.0f; }

#if HAVE_TDM
  const uint32_t ldsK = (uint32_t)(uintptr_t)(void*)&lK[0];
  const uint32_t ldsV = (uint32_t)(uintptr_t)(void*)&lV[0];
#endif

  for (int kv0 = 0; kv0 < SQ; kv0 += 32) {
    __syncthreads();  // all waves done reading previous tiles
    const int rowg0 = h * 128 + (kv0 >> 4);     // (kv0*192)%3072 == 0
    const _Float16* tileBase = qkv + batchOff + (size_t)rowg0 * 3072;
#if HAVE_TDM
    if (wave < 2) {
      // wave 0 -> K (c=64..127), wave 1 -> V (c=128..191)
      tdm_load_tile(wave == 0 ? ldsK : ldsV,
                    tileBase + (wave == 0 ? 64 : 128));
      __builtin_amdgcn_s_wait_tensorcnt(0);
    }
#else
    {   // fallback: cooperative vector-load path (row-major K and V)
      int r = tid >> 3, dc = (tid & 7) * 8;
      const _Float16* kp = tileBase + (size_t)(r >> 4) * 3072 + (r & 15) * 192 + 64 + dc;
      *reinterpret_cast<v8h*>(&lK[r * 72 + dc]) = ld8(kp);
      *reinterpret_cast<v8h*>(&lV[r * 72 + dc]) = ld8(kp + 64);
    }
#endif
    __syncthreads();

    // ---- scores: S = (q/8) @ K^T; all 4 B-fragments first, then 4 WMMAs ----
    const _Float16* kr0 = &lK[l16 * 72];
    const _Float16* kr1 = &lK[(16 + l16) * 72];
    v16h kb00 = cat16(ld8(kr0 + half * 16), ld8(kr0 + half * 16 + 8));
    v16h kb01 = cat16(ld8(kr0 + 32 + half * 16), ld8(kr0 + 40 + half * 16));
    v16h kb10 = cat16(ld8(kr1 + half * 16), ld8(kr1 + half * 16 + 8));
    v16h kb11 = cat16(ld8(kr1 + 32 + half * 16), ld8(kr1 + 40 + half * 16));
    v8f s0 = {}, s1 = {};
    s0 = __builtin_amdgcn_wmma_f32_16x16x32_f16(false, q0, false, kb00, (short)0, s0, false, false);
    s0 = __builtin_amdgcn_wmma_f32_16x16x32_f16(false, q1, false, kb01, (short)0, s0, false, false);
    s1 = __builtin_amdgcn_wmma_f32_16x16x32_f16(false, q0, false, kb10, (short)0, s1, false, false);
    s1 = __builtin_amdgcn_wmma_f32_16x16x32_f16(false, q1, false, kb11, (short)0, s1, false, false);

    // ---- online softmax (DPP16 reductions) ----
    _Float16* lPw = lP[wave];
#pragma unroll
    for (int i = 0; i < 8; ++i) {
      float v0 = s0[i], v1 = s1[i];
      float mx = row16_max(fmaxf(v0, v1));
      float mnew = fmaxf(mrun[i], mx);
      float f = __expf(mrun[i] - mnew);
      float p0 = __expf(v0 - mnew);
      float p1 = __expf(v1 - mnew);
      float rs = row16_sum(p0 + p1);
      lrun[i] = lrun[i] * f + rs;
      mrun[i] = mnew;
      oacc[0][i] *= f; oacc[1][i] *= f; oacc[2][i] *= f; oacc[3][i] *= f;
      int m = i + half * 8;
      lPw[m * 40 + l16] = (_Float16)p0;
      lPw[m * 40 + 16 + l16] = (_Float16)p1;
    }

    // ---- O += P @ V; all fragments first, then 4 WMMAs ----
    const _Float16* pr = &lPw[l16 * 40];
    v16h pa = cat16(ld8(pr + half * 16), ld8(pr + half * 16 + 8));
    v16h vb0, vb1, vb2, vb3;
#pragma unroll
    for (int e = 0; e < 16; ++e) {
      const _Float16* vrow = &lV[(half * 16 + e) * 72 + l16];
      vb0[e] = vrow[0];
      vb1[e] = vrow[16];
      vb2[e] = vrow[32];
      vb3[e] = vrow[48];
    }
    oacc[0] = __builtin_amdgcn_wmma_f32_16x16x32_f16(false, pa, false, vb0, (short)0, oacc[0], false, false);
    oacc[1] = __builtin_amdgcn_wmma_f32_16x16x32_f16(false, pa, false, vb1, (short)0, oacc[1], false, false);
    oacc[2] = __builtin_amdgcn_wmma_f32_16x16x32_f16(false, pa, false, vb2, (short)0, oacc[2], false, false);
    oacc[3] = __builtin_amdgcn_wmma_f32_16x16x32_f16(false, pa, false, vb3, (short)0, oacc[3], false, false);
  }

  // ---- normalize and store O[b, s, h*64 + d] as f16 ----
#pragma unroll
  for (int i = 0; i < 8; ++i) {
    float inv = 1.0f / lrun[i];
    int srow = qbase + i + half * 8;
    size_t rowo = ((size_t)b * SQ + srow) * MD + h * HD;
#pragma unroll
    for (int t = 0; t < 4; ++t) {
      O[rowo + t * 16 + l16] = (_Float16)(oacc[t][i] * inv);
    }
  }
}

extern "C" void kernel_launch(void* const* d_in, const int* in_sizes, int n_in,
                              void* d_out, int out_size, void* d_ws, size_t ws_size,
                              hipStream_t stream) {
  (void)in_sizes; (void)n_in; (void)out_size; (void)ws_size;
  const float* x     = (const float*)d_in[0];
  const float* w_qkv = (const float*)d_in[1];
  const float* b_qkv = (const float*)d_in[2];
  const float* w_out = (const float*)d_in[3];
  const float* b_out = (const float*)d_in[4];
  float* out = (float*)d_out;

  _Float16* qkv_h = (_Float16*)d_ws;                 // (B*S, 3072) f16: 24 MiB
  _Float16* o_h   = qkv_h + (size_t)BA * SQ * 3072;  // (B*S, 1024) f16:  8 MiB

  dim3 blk(256);
  gemm_bias_kernel<float, _Float16>
      <<<dim3(3072 / 64, (BA * SQ) / 128), blk, 0, stream>>>(
          x, w_qkv, b_qkv, qkv_h, BA * SQ, 3 * MD, MD);
  attn_kernel<<<dim3(SQ / 128, NH, BA), blk, 0, stream>>>(qkv_h, o_h);
  gemm_bias_kernel<_Float16, float>
      <<<dim3(MD / 64, (BA * SQ) / 128), blk, 0, stream>>>(
          o_h, w_out, b_out, out, BA * SQ, MD, MD);
}
